// MOE_70557722738901
// MI455X (gfx1250) — matrix operand order repaired and checked by reading, not compiled, not run on hardware
//
#include <hip/hip_runtime.h>
#include <math.h>

#define HIDDEN   2048
#define INTER    1408
#define SINTER   2816   // N_SHARED * INTER
#define NEXP     16
#define TOPK     4
#define T_TOKENS 4096

typedef __bf16 bf16_t;
typedef __attribute__((ext_vector_type(16))) __bf16 v16bf;
typedef __attribute__((ext_vector_type(8)))  float  v8f;

static __device__ __forceinline__ int imin(int a, int b) { return a < b ? a : b; }

// ---------------------------------------------------------------------------
// A fragment: 16x32 bf16, row = lane&15.
// lane<16 : K = k0+0..7 (V0-3) and k0+16..23 (V4-7)
// lane>=16: K = k0+8..15 and k0+24..31
// ---------------------------------------------------------------------------
static __device__ __forceinline__ v16bf load_a_frag(const bf16_t* __restrict__ row,
                                                    int k0, int lane) {
    const bf16_t* p = row + k0 + ((lane & 16) ? 8 : 0);
    union { v16bf v; uint4 u[2]; } a;
    a.u[0] = *(const uint4*)(p);
    a.u[1] = *(const uint4*)(p + 16);
    return a.v;
}

// ---------------------------------------------------------------------------
// B fragment from LDS tile staged as [64 cols][32 K] bf16.
// lane holds column (t*16 + lane&15); lane<16 -> K 0..7,16..23, lane>=16 -> K 8..15,24..31
// ---------------------------------------------------------------------------
static __device__ __forceinline__ v16bf load_b_lds(const bf16_t* __restrict__ lds,
                                                   int t, int lane) {
    const bf16_t* p = lds + ((t * 16 + (lane & 15)) * 32 + ((lane & 16) ? 8 : 0));
    union { v16bf v; uint4 u[2]; } b;
    b.u[0] = *(const uint4*)(p);
    b.u[1] = *(const uint4*)(p + 16);
    return b.v;
}

// ---------------------------------------------------------------------------
// Software-pipelined weight staging: global fp32 tile -> regs -> bf16 LDS.
// 256 threads: col = tid>>2, each thread handles 8 contiguous K (32B load).
// ---------------------------------------------------------------------------
struct StagedTile { float4 f0, f1; };

static __device__ __forceinline__ StagedTile load_w_tile(const float* __restrict__ W,
                                                         int cbase, int K, int k0, int tid) {
    int scol  = tid >> 2;
    int skoff = (tid & 3) * 8;
    const float* g = W + (size_t)(cbase + scol) * K + (k0 + skoff);
    StagedTile s;
    s.f0 = ((const float4*)g)[0];
    s.f1 = ((const float4*)g)[1];
    return s;
}

static __device__ __forceinline__ void store_w_tile(bf16_t* __restrict__ lds,
                                                    const StagedTile& s, int tid) {
    int scol  = tid >> 2;
    int skoff = (tid & 3) * 8;
    union { uint4 u; bf16_t e[8]; } pk;
    pk.e[0] = (bf16_t)s.f0.x; pk.e[1] = (bf16_t)s.f0.y;
    pk.e[2] = (bf16_t)s.f0.z; pk.e[3] = (bf16_t)s.f0.w;
    pk.e[4] = (bf16_t)s.f1.x; pk.e[5] = (bf16_t)s.f1.y;
    pk.e[6] = (bf16_t)s.f1.z; pk.e[7] = (bf16_t)s.f1.w;
    *(uint4*)(lds + scol * 32 + skoff) = pk.u;
}

static __device__ __forceinline__ v8f zero8() {
    v8f z;
#pragma unroll
    for (int i = 0; i < 8; ++i) z[i] = 0.0f;
    return z;
}

// ---------------------------------------------------------------------------
// x fp32 -> bf16
// ---------------------------------------------------------------------------
__global__ __launch_bounds__(256) void convert_x_kernel(const float* __restrict__ x,
                                                        bf16_t* __restrict__ xb, int n) {
    int i = (blockIdx.x * 256 + threadIdx.x) * 4;
    if (i + 3 < n) {
        float4 f = *(const float4*)(x + i);
        union { unsigned long long u; bf16_t e[4]; } pk;
        pk.e[0] = (bf16_t)f.x; pk.e[1] = (bf16_t)f.y;
        pk.e[2] = (bf16_t)f.z; pk.e[3] = (bf16_t)f.w;
        *(unsigned long long*)(xb + i) = pk.u;
    }
}

// ---------------------------------------------------------------------------
// Gate: one wave per token. logits = x @ gate_w.T (fp32), softmax, top-4.
// ---------------------------------------------------------------------------
__global__ __launch_bounds__(256) void gate_kernel(const float* __restrict__ x,
                                                   const float* __restrict__ gw,
                                                   int* __restrict__ topk_idx,
                                                   float* __restrict__ topk_w) {
    int t    = blockIdx.x * 8 + (threadIdx.x >> 5);
    int lane = threadIdx.x & 31;
    const float* xr = x + (size_t)t * HIDDEN;
    float scores[NEXP];
#pragma unroll
    for (int e = 0; e < NEXP; ++e) {
        const float* wr = gw + (size_t)e * HIDDEN;
        float s = 0.f;
        for (int k = lane; k < HIDDEN; k += 32) s += xr[k] * wr[k];
#pragma unroll
        for (int off = 16; off > 0; off >>= 1) s += __shfl_xor(s, off, 32);
        scores[e] = s;  // full butterfly: all lanes hold the result
    }
    float mx = scores[0];
#pragma unroll
    for (int e = 1; e < NEXP; ++e) mx = fmaxf(mx, scores[e]);
    float sum = 0.f;
#pragma unroll
    for (int e = 0; e < NEXP; ++e) { scores[e] = __expf(scores[e] - mx); sum += scores[e]; }
    float inv = 1.f / sum;
#pragma unroll
    for (int e = 0; e < NEXP; ++e) scores[e] *= inv;

    bool used[NEXP];
#pragma unroll
    for (int e = 0; e < NEXP; ++e) used[e] = false;
    for (int k = 0; k < TOPK; ++k) {
        int bi = 0; float bv = -1.f;
        for (int e = 0; e < NEXP; ++e)
            if (!used[e] && scores[e] > bv) { bv = scores[e]; bi = e; }
        used[bi] = true;
        if (lane == 0) {
            topk_idx[t * TOPK + k] = bi;
            topk_w[t * TOPK + k]   = bv;  // ROUTE_SCALE == 1.0
        }
    }
}

// ---------------------------------------------------------------------------
// Deterministic per-expert token lists (one wave per expert, ballot prefix).
// ---------------------------------------------------------------------------
__global__ void build_lists_kernel(const int* __restrict__ topk_idx,
                                   const float* __restrict__ topk_w,
                                   int* __restrict__ counts,
                                   int* __restrict__ token_list,
                                   float* __restrict__ weight_list) {
    int e = blockIdx.x;
    int lane = threadIdx.x;
    int base = 0;
    for (int t0 = 0; t0 < T_TOKENS; t0 += 32) {
        int t = t0 + lane;
        bool hit = false; float w = 0.f;
#pragma unroll
        for (int k = 0; k < TOPK; ++k) {
            if (topk_idx[t * TOPK + k] == e) { hit = true; w = topk_w[t * TOPK + k]; }
        }
        unsigned m = (unsigned)__ballot(hit);
        int pre = __popc(m & ((1u << lane) - 1u));
        if (hit) {
            token_list[(size_t)e * T_TOKENS + base + pre]  = t;
            weight_list[(size_t)e * T_TOKENS + base + pre] = w;
        }
        base += __popc(m);
    }
    if (lane == 0) counts[e] = base;
}

__global__ void offsets_kernel(const int* __restrict__ counts, int* __restrict__ offsets) {
    if (threadIdx.x == 0 && blockIdx.x == 0) {
        int s = 0;
        for (int e = 0; e < NEXP; ++e) { offsets[e] = s; s += counts[e]; }
    }
}

// ---------------------------------------------------------------------------
// GEMM1 + fused SwiGLU: h = silu(x@W1.T + b1) * (x@W3.T + b3), bf16 out.
// Block: 128 rows x 64 cols, 8 waves (wave = 16-row strip, 4 WMMA col tiles).
// Pipelined: next K-tile's fp32 weights prefetched into regs during WMMAs.
// token_list==nullptr -> shared expert (identity rows, plain indexing).
// ---------------------------------------------------------------------------
__global__ __launch_bounds__(256) void gemm1_swiglu_kernel(
    const bf16_t* __restrict__ X,            // [*, HIDDEN] bf16
    const int* __restrict__ token_list,      // [E, T] or nullptr
    const int* __restrict__ counts,          // [E] or nullptr
    const int* __restrict__ offsets,         // [E] or nullptr
    int fixedM,
    const float* __restrict__ W1, const float* __restrict__ B1,
    const float* __restrict__ W3, const float* __restrict__ B3,
    int N, int K,
    bf16_t* __restrict__ H)                  // [M, N] bf16 rows at slot offset
{
    __shared__ bf16_t lds1[64 * 32];
    __shared__ bf16_t lds3[64 * 32];

    int e = blockIdx.z;
    int M = counts ? counts[e] : fixedM;
    int mbase = blockIdx.y * 128;
    if (mbase >= M) return;
    int cbase = blockIdx.x * 64;

    int tid  = threadIdx.x;
    int wave = tid >> 5;
    int lane = tid & 31;
    int slot_base = offsets ? offsets[e] : 0;

    int arow   = mbase + wave * 16 + (lane & 15);
    int arow_c = imin(arow, M - 1);
    int src    = token_list ? token_list[(size_t)e * T_TOKENS + arow_c] : arow_c;
    const bf16_t* arow_ptr = X + (size_t)src * HIDDEN;

    const float* W1e = W1 + (size_t)e * N * K;
    const float* W3e = W3 + (size_t)e * N * K;
    const float* B1e = B1 + (size_t)e * N;
    const float* B3e = B3 + (size_t)e * N;

    v8f acc1[4], acc3[4];
#pragma unroll
    for (int t = 0; t < 4; ++t) { acc1[t] = zero8(); acc3[t] = zero8(); }

    // prologue: fetch first K-tile into registers
    StagedTile s1 = load_w_tile(W1e, cbase, K, 0, tid);
    StagedTile s3 = load_w_tile(W3e, cbase, K, 0, tid);

    for (int k0 = 0; k0 < K; k0 += 32) {
        __syncthreads();                     // previous tile's LDS reads done
        store_w_tile(lds1, s1, tid);
        store_w_tile(lds3, s3, tid);
        __syncthreads();                     // tile visible to all waves

        if (k0 + 32 < K) {                   // prefetch next tile (hidden by WMMAs)
            s1 = load_w_tile(W1e, cbase, K, k0 + 32, tid);
            s3 = load_w_tile(W3e, cbase, K, k0 + 32, tid);
        }

        v16bf a = load_a_frag(arow_ptr, k0, lane);
        v16bf b1f[4], b3f[4];
#pragma unroll
        for (int t = 0; t < 4; ++t) {
            b1f[t] = load_b_lds(lds1, t, lane);
            b3f[t] = load_b_lds(lds3, t, lane);
        }
#pragma unroll
        for (int t = 0; t < 4; ++t) {
            acc1[t] = __builtin_amdgcn_wmma_f32_16x16x32_bf16(
                false, a, false, b1f[t], (short)0, acc1[t], false, false);
            acc3[t] = __builtin_amdgcn_wmma_f32_16x16x32_bf16(
                false, a, false, b3f[t], (short)0, acc3[t], false, false);
        }
    }

    int rbase = mbase + wave * 16 + ((lane & 16) ? 8 : 0);
#pragma unroll
    for (int t = 0; t < 4; ++t) {
        int col = cbase + t * 16 + (lane & 15);
        float b1v = B1e[col];
        float b3v = B3e[col];
#pragma unroll
        for (int d = 0; d < 8; ++d) {
            int r = rbase + d;
            if (r < M) {
                float y1 = acc1[t][d] + b1v;
                float y3 = acc3[t][d] + b3v;
                float hv = (y1 / (1.f + __expf(-y1))) * y3;   // silu(y1) * y3
                H[(size_t)(slot_base + r) * N + col] = (bf16_t)hv;
            }
        }
    }
}

// ---------------------------------------------------------------------------
// GEMM2: out_row += gw * (h @ W2.T + b2)  (routed: atomic scatter)
//        out_row  =       h @ W2.T + b2   (shared: plain store, initializes out)
// N fixed = HIDDEN (2048).
// ---------------------------------------------------------------------------
__global__ __launch_bounds__(256) void gemm2_kernel(
    const bf16_t* __restrict__ Hbuf,         // [M, K] bf16 rows at slot offset
    const int* __restrict__ token_list,      // [E, T] or nullptr
    const float* __restrict__ weight_list,   // [E, T] or nullptr
    const int* __restrict__ counts,
    const int* __restrict__ offsets,
    int fixedM,
    const float* __restrict__ W2, const float* __restrict__ B2,
    int K,
    float* __restrict__ out)
{
    __shared__ bf16_t lds2[64 * 32];

    int e = blockIdx.z;
    int M = counts ? counts[e] : fixedM;
    int mbase = blockIdx.y * 128;
    if (mbase >= M) return;
    int cbase = blockIdx.x * 64;

    int tid  = threadIdx.x;
    int wave = tid >> 5;
    int lane = tid & 31;
    int slot_base = offsets ? offsets[e] : 0;

    int arow   = mbase + wave * 16 + (lane & 15);
    int arow_c = imin(arow, M - 1);
    const bf16_t* arow_ptr = Hbuf + (size_t)(slot_base + arow_c) * K;

    const float* W2e = W2 + (size_t)e * HIDDEN * K;
    const float* B2e = B2 + (size_t)e * HIDDEN;

    v8f acc[4];
#pragma unroll
    for (int t = 0; t < 4; ++t) acc[t] = zero8();

    StagedTile s2 = load_w_tile(W2e, cbase, K, 0, tid);

    for (int k0 = 0; k0 < K; k0 += 32) {
        __syncthreads();
        store_w_tile(lds2, s2, tid);
        __syncthreads();

        if (k0 + 32 < K) {
            s2 = load_w_tile(W2e, cbase, K, k0 + 32, tid);
        }

        v16bf a = load_a_frag(arow_ptr, k0, lane);
        v16bf bf[4];
#pragma unroll
        for (int t = 0; t < 4; ++t) bf[t] = load_b_lds(lds2, t, lane);
#pragma unroll
        for (int t = 0; t < 4; ++t) {
            acc[t] = __builtin_amdgcn_wmma_f32_16x16x32_bf16(
                false, a, false, bf[t], (short)0, acc[t], false, false);
        }
    }

    int rbase = mbase + wave * 16 + ((lane & 16) ? 8 : 0);
#pragma unroll
    for (int t = 0; t < 4; ++t) {
        int col = cbase + t * 16 + (lane & 15);
        float b2v = B2e[col];
#pragma unroll
        for (int d = 0; d < 8; ++d) {
            int r = rbase + d;
            if (r < M) {
                float v = acc[t][d] + b2v;
                if (token_list) {
                    int   tok = token_list[(size_t)e * T_TOKENS + r];
                    float gwv = weight_list[(size_t)e * T_TOKENS + r];
                    atomicAdd(&out[(size_t)tok * HIDDEN + col], v * gwv);
                } else {
                    out[(size_t)r * HIDDEN + col] = v;
                }
            }
        }
    }
}

// ---------------------------------------------------------------------------
// Host launcher
// ---------------------------------------------------------------------------
static size_t align256(size_t x) { return (x + 255) & ~(size_t)255; }

extern "C" void kernel_launch(void* const* d_in, const int* in_sizes, int n_in,
                              void* d_out, int out_size, void* d_ws, size_t ws_size,
                              hipStream_t stream) {
    const float* x    = (const float*)d_in[0];   // [2,2048,2048]
    const float* gw   = (const float*)d_in[1];   // [16,2048]
    const float* w1   = (const float*)d_in[2];   // [16,1408,2048]
    const float* b1   = (const float*)d_in[3];   // [16,1408]
    const float* w2   = (const float*)d_in[4];   // [16,2048,1408]
    const float* b2   = (const float*)d_in[5];   // [16,2048]
    const float* w3   = (const float*)d_in[6];   // [16,1408,2048]
    const float* b3   = (const float*)d_in[7];   // [16,1408]
    const float* sw1  = (const float*)d_in[8];   // [2816,2048]
    const float* sb1  = (const float*)d_in[9];   // [2816]
    const float* sw2  = (const float*)d_in[10];  // [2048,2816]
    const float* sb2  = (const float*)d_in[11];  // [2048]
    const float* sw3  = (const float*)d_in[12];  // [2816,2048]
    const float* sb3  = (const float*)d_in[13];  // [2816]
    float* out = (float*)d_out;                  // [2,2048,2048] fp32

    char* ws = (char*)d_ws;
    size_t off = 0;
    int*    topk_idx    = (int*)(ws + off);    off = align256(off + (size_t)T_TOKENS * TOPK * 4);
    float*  topk_w      = (float*)(ws + off);  off = align256(off + (size_t)T_TOKENS * TOPK * 4);
    int*    counts      = (int*)(ws + off);    off = align256(off + NEXP * 4);
    int*    offsets     = (int*)(ws + off);    off = align256(off + NEXP * 4);
    int*    token_list  = (int*)(ws + off);    off = align256(off + (size_t)NEXP * T_TOKENS * 4);
    float*  weight_list = (float*)(ws + off);  off = align256(off + (size_t)NEXP * T_TOKENS * 4);
    bf16_t* x_bf16      = (bf16_t*)(ws + off); off = align256(off + (size_t)T_TOKENS * HIDDEN * 2);
    // h buffer: routed (16384 x 1408) and shared (4096 x 2816) do not overlap in time (shared runs first)
    bf16_t* h_buf       = (bf16_t*)(ws + off);
    // routed need = 16384*1408*2 = 46.1 MB (> shared 23.1 MB); total ws ~64 MB

    // 1) activations fp32 -> bf16
    convert_x_kernel<<<(T_TOKENS * HIDDEN) / (256 * 4), 256, 0, stream>>>(
        x, x_bf16, T_TOKENS * HIDDEN);

    // 2) gate + routing
    gate_kernel<<<T_TOKENS / 8, 256, 0, stream>>>(x, gw, topk_idx, topk_w);
    build_lists_kernel<<<NEXP, 32, 0, stream>>>(topk_idx, topk_w, counts,
                                                token_list, weight_list);
    offsets_kernel<<<1, 32, 0, stream>>>(counts, offsets);

    // 3) shared expert (writes full d_out, initializing it)
    {
        dim3 g1(SINTER / 64, T_TOKENS / 128, 1);
        gemm1_swiglu_kernel<<<g1, 256, 0, stream>>>(
            x_bf16, nullptr, nullptr, nullptr, T_TOKENS,
            sw1, sb1, sw3, sb3, SINTER, HIDDEN, h_buf);
        dim3 g2(HIDDEN / 64, T_TOKENS / 128, 1);
        gemm2_kernel<<<g2, 256, 0, stream>>>(
            h_buf, nullptr, nullptr, nullptr, nullptr, T_TOKENS,
            sw2, sb2, SINTER, out);
    }

    // 4) routed experts (atomic accumulate into d_out)
    {
        dim3 g1(INTER / 64, T_TOKENS / 128, NEXP);
        gemm1_swiglu_kernel<<<g1, 256, 0, stream>>>(
            x_bf16, token_list, counts, offsets, 0,
            w1, b1, w3, b3, INTER, HIDDEN, h_buf);
        dim3 g2(HIDDEN / 64, T_TOKENS / 128, NEXP);
        gemm2_kernel<<<g2, 256, 0, stream>>>(
            h_buf, token_list, weight_list, counts, offsets, 0,
            w2, b2, INTER, out);
    }
}
